// VQVAETimeSeries_16509854286431
// MI455X (gfx1250) — matrix operand order repaired and checked by reading, not compiled, not run on hardware
//
#include <hip/hip_runtime.h>
#include <hip/hip_fp16.h>

// ---------------------------------------------------------------------------
// VQ-VAE forward for MI455X (gfx1250, wave32, WMMA).
// GEMMs use V_WMMA_F32_16X16X32_F16. Weights are pre-transposed to [N][K] so
// every A/B fragment is built from contiguous ds_load_b128 pairs; LDS rows are
// padded to avoid bank conflicts; staging uses CDNA5 async-to-LDS loads.
// ---------------------------------------------------------------------------

typedef __attribute__((ext_vector_type(16))) _Float16 v16h;
typedef __attribute__((ext_vector_type(8)))  _Float16 v8h;
typedef __attribute__((ext_vector_type(8)))  float    v8f;
typedef __attribute__((ext_vector_type(4)))  int      v4i;

constexpr int BATCH  = 16384;
constexpr int TD     = 512;   // T*D
constexpr int HID    = 128;
constexpr int LAT    = 512;   // N_TOKENS * CODE_DIM
constexpr int NCODES = 512;
constexpr int CDIM   = 64;
constexpr int NTOK   = 8;
constexpr long TOKENS = (long)BATCH * NTOK;   // 131072

#if defined(__gfx1250__) && __has_builtin(__builtin_amdgcn_global_load_async_to_lds_b128)
#define HAVE_ASYNC 1
#else
#define HAVE_ASYNC 0
#endif

#if HAVE_ASYNC
typedef __attribute__((address_space(1))) v4i gv4i;   // global 16B vector
typedef __attribute__((address_space(3))) v4i lv4i;   // LDS 16B vector
__device__ __forceinline__ void async_copy_b128(const void* g, void* l) {
  __builtin_amdgcn_global_load_async_to_lds_b128((gv4i*)g, (lv4i*)l, 0, 0);
}
__device__ __forceinline__ void async_wait0() {
#if __has_builtin(__builtin_amdgcn_s_wait_asynccnt)
  __builtin_amdgcn_s_wait_asynccnt(0);
#else
  asm volatile("s_wait_asynccnt 0" ::: "memory");
#endif
}
#endif

// ---------------------------------------------------------------------------
// Tiled WMMA GEMM:  out[M,N] = act(A[M,K] @ W[K,N] + bias),  Wt = W^T [N][K].
// Block = 256 threads = 8 waves. Block tile: 32 rows x N cols.
// wave w -> (m_sub = w&1) selects 16 rows, (strip = w>>1) selects N/4 cols.
// Wt streamed through LDS in 32-K chunks; one WMMA K-step per chunk.
// ---------------------------------------------------------------------------
template <int K, int N, bool RELU, bool A_F16, bool OUT_F16, bool OUT_F32>
__global__ __launch_bounds__(256) void wmma_gemm(
    const void* __restrict__ Av, const _Float16* __restrict__ Wt,
    const float* __restrict__ bias,
    _Float16* __restrict__ out16, float* __restrict__ out32)
{
  constexpr int NSTRIP = N / 4;
  constexpr int NT     = NSTRIP / 16;
  constexpr int SA     = K + 8;    // padded A row (halves): lane stride 4 banks
  constexpr int SW     = 40;       // padded W row (32 + 8): stride 20 banks

  __shared__ alignas(16) _Float16 sA[32 * SA];
  __shared__ alignas(16) _Float16 sW[N * SW];

  const int  tid   = threadIdx.x;
  const int  lane  = tid & 31;
  const int  wave  = tid >> 5;
  const int  m_sub = wave & 1;
  const int  strip = wave >> 1;
  const long mBase = (long)blockIdx.x * 32;

  // ---- stage A tile (32 x K) with padded rows ----
  {
    const long base = mBase * (long)K;
    if constexpr (A_F16) {
      constexpr int AR = K / 8;                   // b128 segments per row
      const _Float16* Ah = (const _Float16*)Av;
#if HAVE_ASYNC
      for (int i = tid; i < 32 * AR; i += 256) {
        const int r = i / AR, seg = (i % AR) * 8;
        async_copy_b128(Ah + base + (long)r * K + seg, sA + r * SA + seg);
      }
#else
      for (int i = tid; i < 32 * AR; i += 256) {
        const int r = i / AR, seg = (i % AR) * 8;
        *(v8h*)(sA + r * SA + seg) = *(const v8h*)(Ah + base + (long)r * K + seg);
      }
#endif
    } else {
      const float* Af = (const float*)Av;
      for (int i = tid; i < 32 * K; i += 256) {
        const int r = i / K, c = i % K;
        sA[r * SA + c] = (_Float16)Af[base + i];
      }
    }
  }

  v8f acc[NT] = {};
  const int nbase0 = strip * NSTRIP;
  const int krow   = (lane < 16) ? 0 : 8;   // B frag: K base select
  const int kb     = (lane < 16) ? 0 : 8;   // A frag: K base select
  const int coll   = lane & 15;

  for (int kc = 0; kc < K / 32; ++kc) {
    // ---- stage Wt chunk: N rows x 32 halves (padded to SW) ----
#if HAVE_ASYNC
    for (int i = tid; i < N * 4; i += 256) {
      const int n = i >> 2, seg = (i & 3) * 8;
      async_copy_b128(Wt + (long)n * K + kc * 32 + seg, sW + n * SW + seg);
    }
    async_wait0();
#else
    for (int i = tid; i < N * 4; i += 256) {
      const int n = i >> 2, seg = (i & 3) * 8;
      *(v8h*)(sW + n * SW + seg) = *(const v8h*)(Wt + (long)n * K + kc * 32 + seg);
    }
#endif
    __syncthreads();

    // ---- A fragment (ISA 16-bit A 16x32 layout): two b128 LDS loads ----
    v16h a;
    {
      const _Float16* row = sA + (m_sub * 16 + coll) * SA + kc * 32;
      ((v8h*)&a)[0] = *(const v8h*)(row + kb);        // K = kb..kb+7
      ((v8h*)&a)[1] = *(const v8h*)(row + kb + 16);   // K = kb+16..kb+23
    }

#pragma unroll
    for (int nt = 0; nt < NT; ++nt) {
      const int col = nbase0 + nt * 16 + coll;
      // B 32x16: lane = col, halves = K krow..krow+7, krow+16..krow+23
      const _Float16* brow = sW + col * SW + krow;
      v16h b;
      ((v8h*)&b)[0] = *(const v8h*)(brow);
      ((v8h*)&b)[1] = *(const v8h*)(brow + 16);
      acc[nt] = __builtin_amdgcn_wmma_f32_16x16x32_f16(
          false, a, false, b, (short)0, acc[nt], false, false);
    }
    __syncthreads();
  }

  // ---- epilogue: D layout lane=col, VGPR r -> M = r + (lane<16?0:8) ----
  const int rbase = (lane < 16) ? 0 : 8;
#pragma unroll
  for (int nt = 0; nt < NT; ++nt) {
    const int   col = nbase0 + nt * 16 + coll;
    const float bv  = bias[col];
#pragma unroll
    for (int r = 0; r < 8; ++r) {
      float v = acc[nt][r] + bv;
      if constexpr (RELU) v = fmaxf(v, 0.0f);
      const long row = mBase + m_sub * 16 + rbase + r;
      if constexpr (OUT_F32) out32[row * N + col] = v;
      if constexpr (OUT_F16) out16[row * N + col] = (_Float16)v;
    }
  }
}

// ---------------------------------------------------------------------------
// LayerNorm over each 64-wide token: one wave per token (2 elems/lane).
// ---------------------------------------------------------------------------
__global__ __launch_bounds__(256) void layernorm_kernel(
    const float* __restrict__ ze32, _Float16* __restrict__ ze16)
{
  const int  lane = threadIdx.x & 31;
  const int  wave = threadIdx.x >> 5;
  const long tok  = (long)blockIdx.x * 8 + wave;
  const float* src = ze32 + tok * 64;
  float x0 = src[lane], x1 = src[lane + 32];
  float s  = x0 + x1;
  float sq = x0 * x0 + x1 * x1;
#pragma unroll
  for (int off = 16; off >= 1; off >>= 1) {
    s  += __shfl_xor(s, off, 32);
    sq += __shfl_xor(sq, off, 32);
  }
  const float mu  = s * (1.0f / 64.0f);
  const float var = sq * (1.0f / 64.0f) - mu * mu;
  const float rs  = rsqrtf(var + 1e-5f);
  _Float16* dst = ze16 + tok * 64;
  dst[lane]      = (_Float16)((x0 - mu) * rs);
  dst[lane + 32] = (_Float16)((x1 - mu) * rs);
}

// ---------------------------------------------------------------------------
// VQ argmin via WMMA: argmin ||z-c||^2 == argmax (z.c - 0.5*||c||^2).
// 8 waves/block, one 16-token tile per wave; f16 codebook [code][64] staged
// in LDS (two halves of 256 codes, rows padded to 72 halves).
// ---------------------------------------------------------------------------
__global__ __launch_bounds__(256) void vq_kernel(
    const _Float16* __restrict__ ze,   // [TOKENS][64] (layernormed, f16)
    const _Float16* __restrict__ cbh,  // [512][64] codebook f16
    const float* __restrict__ c2,      // [512] ||c||^2
    float* __restrict__ idx_f, int* __restrict__ idx_i)
{
  constexpr int SC = 72;               // padded codebook row (halves)
  __shared__ alignas(16) _Float16 sCB[256 * SC];

  const int  tid     = threadIdx.x;
  const int  lane    = tid & 31;
  const int  wave    = tid >> 5;
  const long tokBase = (long)blockIdx.x * 128 + wave * 16;

  // A fragments for 16 tokens, K=64 split into two 32-chunks
  v16h a0, a1;
  {
    const long tok = tokBase + (lane & 15);
    const _Float16* row = ze + tok * 64;
    const int kb = (lane < 16) ? 0 : 8;
    ((v8h*)&a0)[0] = *(const v8h*)(row + kb);
    ((v8h*)&a0)[1] = *(const v8h*)(row + kb + 16);
    ((v8h*)&a1)[0] = *(const v8h*)(row + 32 + kb);
    ((v8h*)&a1)[1] = *(const v8h*)(row + 32 + kb + 16);
  }

  float best[8];
  int   bidx[8];
#pragma unroll
  for (int r = 0; r < 8; ++r) { best[r] = -3.0e38f; bidx[r] = 0; }

  const int krow = (lane < 16) ? 0 : 8;
  const int coll = lane & 15;

  for (int half = 0; half < 2; ++half) {
    // stage 256 codebook rows (64 halves each -> 8 b128 per row)
#if HAVE_ASYNC
    for (int i = tid; i < 256 * 8; i += 256) {
      const int c = i >> 3, seg = (i & 7) * 8;
      async_copy_b128(cbh + (long)(half * 256 + c) * 64 + seg, sCB + c * SC + seg);
    }
    async_wait0();
#else
    for (int i = tid; i < 256 * 8; i += 256) {
      const int c = i >> 3, seg = (i & 7) * 8;
      *(v8h*)(sCB + c * SC + seg) =
          *(const v8h*)(cbh + (long)(half * 256 + c) * 64 + seg);
    }
#endif
    __syncthreads();

    for (int t = 0; t < 16; ++t) {
      const int c = t * 16 + coll;
      const _Float16* crow = sCB + c * SC;
      v16h b0, b1;                       // K krow..+7, krow+16..+23 (and +32)
      ((v8h*)&b0)[0] = *(const v8h*)(crow + krow);
      ((v8h*)&b0)[1] = *(const v8h*)(crow + krow + 16);
      ((v8h*)&b1)[0] = *(const v8h*)(crow + 32 + krow);
      ((v8h*)&b1)[1] = *(const v8h*)(crow + 32 + krow + 16);
      v8f accv = {};
      accv = __builtin_amdgcn_wmma_f32_16x16x32_f16(false, a0, false, b0,
                                                    (short)0, accv, false, false);
      accv = __builtin_amdgcn_wmma_f32_16x16x32_f16(false, a1, false, b1,
                                                    (short)0, accv, false, false);
      const int   code    = half * 256 + c;
      const float half_c2 = 0.5f * c2[code];
#pragma unroll
      for (int r = 0; r < 8; ++r) {
        const float sc = accv[r] - half_c2;
        if (sc > best[r]) { best[r] = sc; bidx[r] = code; }
      }
    }
    __syncthreads();
  }

  // argmax reduce across the 16 lanes that share each token row
#pragma unroll
  for (int r = 0; r < 8; ++r) {
#pragma unroll
    for (int off = 8; off >= 1; off >>= 1) {
      const float ob = __shfl_xor(best[r], off, 32);
      const int   oi = __shfl_xor(bidx[r], off, 32);
      if (ob > best[r] || (ob == best[r] && oi < bidx[r])) {
        best[r] = ob; bidx[r] = oi;
      }
    }
  }
  if ((lane & 15) == 0) {
    const long tok0 = tokBase + ((lane < 16) ? 0 : 8);
#pragma unroll
    for (int r = 0; r < 8; ++r) {
      idx_f[tok0 + r] = (float)bidx[r];
      idx_i[tok0 + r] = bidx[r];
    }
  }
}

// ---------------------------------------------------------------------------
// Small helper kernels
// ---------------------------------------------------------------------------
__global__ __launch_bounds__(256) void cvt_kernel(const float* __restrict__ s,
                                                  _Float16* __restrict__ d, int n) {
  int i = blockIdx.x * 256 + threadIdx.x;
  if (i < n) d[i] = (_Float16)s[i];
}

// f32 W[K][N] -> f16 Wt[N][K]
__global__ __launch_bounds__(256) void cvtT_kernel(const float* __restrict__ s,
                                                   _Float16* __restrict__ d,
                                                   int K, int N) {
  int i = blockIdx.x * 256 + threadIdx.x;
  if (i < K * N) {
    const int k = i / N, n = i % N;     // coalesced reads
    d[(long)n * K + k] = (_Float16)s[i];
  }
}

__global__ __launch_bounds__(256) void c2_kernel(const float* __restrict__ cb,
                                                 float* __restrict__ c2) {
  int k = blockIdx.x * 256 + threadIdx.x;
  if (k < NCODES) {
    float s = 0.f;
    for (int d = 0; d < CDIM; ++d) { float v = cb[k * CDIM + d]; s += v * v; }
    c2[k] = s;
  }
}

__global__ __launch_bounds__(256) void gather_kernel(
    const float* __restrict__ cb, const int* __restrict__ idx,
    float* __restrict__ zq32, _Float16* __restrict__ zq16) {
  const long g   = (long)blockIdx.x * 256 + threadIdx.x;  // TOKENS*64
  const long tok = g >> 6;
  const int  d   = (int)(g & 63);
  const int  k   = idx[tok];
  const float v  = cb[(long)k * CDIM + d];
  zq32[g] = v;
  zq16[g] = (_Float16)v;
}

// ---------------------------------------------------------------------------
// Host launcher
// ---------------------------------------------------------------------------
extern "C" void kernel_launch(void* const* d_in, const int* in_sizes, int n_in,
                              void* d_out, int out_size, void* d_ws, size_t ws_size,
                              hipStream_t stream)
{
  (void)in_sizes; (void)n_in; (void)out_size; (void)ws_size;
  const float* x      = (const float*)d_in[0];
  const float* enc_w1 = (const float*)d_in[1];
  const float* enc_b1 = (const float*)d_in[2];
  const float* enc_w2 = (const float*)d_in[3];
  const float* enc_b2 = (const float*)d_in[4];
  const float* cb     = (const float*)d_in[5];
  const float* dec_w1 = (const float*)d_in[6];
  const float* dec_b1 = (const float*)d_in[7];
  const float* dec_w2 = (const float*)d_in[8];
  const float* dec_b2 = (const float*)d_in[9];

  float* recon   = (float*)d_out;                          // [B*T*D]
  float* zq_out  = recon  + (size_t)BATCH * TD;            // [TOKENS*64]
  float* idx_out = zq_out + (size_t)BATCH * LAT;           // [TOKENS]

  char* p = (char*)d_ws;
  auto carve = [&](size_t bytes) {
    char* q = p; p += (bytes + 255) & ~(size_t)255; return q;
  };
  _Float16* w1t  = (_Float16*)carve((size_t)TD * HID * 2);   // [HID][TD]
  _Float16* w2t  = (_Float16*)carve((size_t)HID * LAT * 2);  // [LAT][HID]
  _Float16* dw1t = (_Float16*)carve((size_t)LAT * HID * 2);  // [HID][LAT]
  _Float16* dw2t = (_Float16*)carve((size_t)HID * TD * 2);   // [TD][HID]
  _Float16* cbh  = (_Float16*)carve((size_t)NCODES * CDIM * 2);
  float*    c2   = (float*)   carve((size_t)NCODES * 4);
  _Float16* h1   = (_Float16*)carve((size_t)BATCH * HID * 2);
  float*    ze32 = (float*)   carve((size_t)BATCH * LAT * 4);
  _Float16* ze16 = (_Float16*)carve((size_t)BATCH * LAT * 2);
  int*      idxi = (int*)     carve((size_t)TOKENS * 4);
  _Float16* zq16 = (_Float16*)carve((size_t)BATCH * LAT * 2);
  _Float16* h2   = (_Float16*)carve((size_t)BATCH * HID * 2);

  // weight / codebook preprocessing (tiny)
  cvtT_kernel<<<(TD * HID + 255) / 256, 256, 0, stream>>>(enc_w1, w1t, TD, HID);
  cvtT_kernel<<<(HID * LAT + 255) / 256, 256, 0, stream>>>(enc_w2, w2t, HID, LAT);
  cvtT_kernel<<<(LAT * HID + 255) / 256, 256, 0, stream>>>(dec_w1, dw1t, LAT, HID);
  cvtT_kernel<<<(HID * TD + 255) / 256, 256, 0, stream>>>(dec_w2, dw2t, HID, TD);
  cvt_kernel<<<(NCODES * CDIM + 255) / 256, 256, 0, stream>>>(cb, cbh, NCODES * CDIM);
  c2_kernel<<<(NCODES + 255) / 256, 256, 0, stream>>>(cb, c2);

  // encoder
  wmma_gemm<TD, HID, true, false, true, false>
      <<<BATCH / 32, 256, 0, stream>>>(x, w1t, enc_b1, h1, nullptr);
  wmma_gemm<HID, LAT, false, true, false, true>
      <<<BATCH / 32, 256, 0, stream>>>(h1, w2t, enc_b2, nullptr, ze32);
  layernorm_kernel<<<(int)(TOKENS / 8), 256, 0, stream>>>(ze32, ze16);

  // vector quantization + gather
  vq_kernel<<<(int)(TOKENS / 128), 256, 0, stream>>>(ze16, cbh, c2, idx_out, idxi);
  gather_kernel<<<(int)(TOKENS * CDIM / 256), 256, 0, stream>>>(cb, idxi, zq_out, zq16);

  // decoder
  wmma_gemm<LAT, HID, true, true, true, false>
      <<<BATCH / 32, 256, 0, stream>>>(zq16, dw1t, dec_b1, h2, nullptr);
  wmma_gemm<HID, TD, false, true, false, true>
      <<<BATCH / 32, 256, 0, stream>>>(h2, dw2t, dec_b2, nullptr, recon);
}
